// GATNet_15891378995747
// MI455X (gfx1250) — matrix operand order repaired and checked by reading, not compiled, not run on hardware
//
#include <hip/hip_runtime.h>

#define N_NODES_C  50000
#define N_EDGES_C  800000
#define N_GRAPHS_C 64

typedef __bf16 v16bf __attribute__((ext_vector_type(16)));
typedef __bf16 v8bf  __attribute__((ext_vector_type(8)));
typedef float  v8f   __attribute__((ext_vector_type(8)));

__device__ __forceinline__ unsigned short f2bfu(float f) {
    unsigned u = __float_as_uint(f);
    u += 0x7FFFu + ((u >> 16) & 1u);            // round-to-nearest-even
    return (unsigned short)(u >> 16);
}

__device__ __forceinline__ void atomicMaxF(float* addr, float val) {
    if (!(val < 0.0f))
        atomicMax((int*)addr, __float_as_int(val));
    else
        atomicMin((unsigned int*)addr, __float_as_uint(val));
}

// ---------------------------------------------------------------------------
// Weight prep: fp32 [K][N] row-major  ->  bf16 bits [N][K] (transposed)
// ---------------------------------------------------------------------------
__global__ void transpose_bf16(const float* __restrict__ W,
                               unsigned short* __restrict__ Wt, int K, int N) {
    int i = blockIdx.x * blockDim.x + threadIdx.x;
    if (i >= K * N) return;
    int k = i / N;
    int n = i - k * N;
    Wt[(size_t)n * K + k] = f2bfu(W[i]);
}

// ---------------------------------------------------------------------------
// GEMM: Out[M][NCOL] = A[M][K] (fp32, cast to bf16) x Wt^T (Wt is [NCOL][K] bf16)
// One block = 16 rows x NCOL cols; NCOL/32 waves, each wave owns two 16x16 tiles.
// ---------------------------------------------------------------------------
template <int K, int NCOL>
__global__ __launch_bounds__(NCOL) void gemm_bf16_wmma(
    const float* __restrict__ A, const unsigned short* __restrict__ Wt,
    float* __restrict__ Out) {
    constexpr int LDA = K + 8;                       // pad to dodge bank conflicts
    __shared__ __align__(16) unsigned short As[16 * LDA];
    const int tid  = threadIdx.x;
    const int row0 = blockIdx.x * 16;

    for (int i = tid; i < 16 * K; i += NCOL) {       // stage A tile as bf16
        const int r = i / K;
        const int k = i - r * K;
        As[r * LDA + k] = f2bfu(A[(size_t)(row0 + r) * K + k]);
    }
    __syncthreads();

    const int lane = tid & 31;
    const int wave = tid >> 5;
    const int lh   = lane & 15;
    const int kha  = (lane >> 4) << 4;               // A half-select: 0 or 16
    const int khb  = (lane >> 4) << 4;               // B half-select: 0 or 16
    const int n0   = wave * 32;

    const unsigned short* ap = &As[lh * LDA + kha];
    const unsigned short* w0 = Wt + (size_t)(n0 + lh) * K + khb;
    const unsigned short* w1 = Wt + (size_t)(n0 + 16 + lh) * K + khb;

    v8f acc0 = {0.f, 0.f, 0.f, 0.f, 0.f, 0.f, 0.f, 0.f};
    v8f acc1 = acc0;
#pragma unroll
    for (int k0 = 0; k0 < K; k0 += 32) {
        v8bf alo = *(const v8bf*)(const void*)(ap + k0);
        v8bf ahi = *(const v8bf*)(const void*)(ap + k0 + 8);
        v16bf a  = __builtin_shufflevector(alo, ahi, 0,1,2,3,4,5,6,7,8,9,10,11,12,13,14,15);
        v8bf b0l = *(const v8bf*)(const void*)(w0 + k0);
        v8bf b0h = *(const v8bf*)(const void*)(w0 + k0 + 8);
        v16bf b0 = __builtin_shufflevector(b0l, b0h, 0,1,2,3,4,5,6,7,8,9,10,11,12,13,14,15);
        v8bf b1l = *(const v8bf*)(const void*)(w1 + k0);
        v8bf b1h = *(const v8bf*)(const void*)(w1 + k0 + 8);
        v16bf b1 = __builtin_shufflevector(b1l, b1h, 0,1,2,3,4,5,6,7,8,9,10,11,12,13,14,15);
        acc0 = __builtin_amdgcn_wmma_f32_16x16x32_bf16(false, a, false, b0,
                                                       (short)0, acc0, false, false);
        acc1 = __builtin_amdgcn_wmma_f32_16x16x32_bf16(false, a, false, b1,
                                                       (short)0, acc1, false, false);
    }

    const int mb = (lane >> 4) << 3;                 // C/D: 0 or 8 row base
#pragma unroll
    for (int r = 0; r < 8; ++r) {
        Out[(size_t)(row0 + mb + r) * NCOL + n0 + lh]      = acc0[r];
        Out[(size_t)(row0 + mb + r) * NCOL + n0 + 16 + lh] = acc1[r];
    }
}

// ---------------------------------------------------------------------------
// Edge phase
// ---------------------------------------------------------------------------
template <int H>
__global__ void edge_logits(const float4* __restrict__ xl, const float4* __restrict__ xr,
                            const float4* __restrict__ att,
                            const int* __restrict__ src, const int* __restrict__ dst,
                            float* __restrict__ logits, float* __restrict__ nmax) {
    const int gid = blockIdx.x * blockDim.x + threadIdx.x;
    if (gid >= N_EDGES_C * H) return;
    const int e = gid / H;
    const int h = gid - e * H;
    const int s = src[e];
    const int d = dst[e];
    const float4* xs = xl + (size_t)s * (H * 16) + h * 16;
    const float4* xd = xr + (size_t)d * (H * 16) + h * 16;
    float acc = 0.f;
#pragma unroll
    for (int c = 0; c < 16; ++c) {
        float4 a = xs[c], b = xd[c], w = att[h * 16 + c];
        float v;
        v = a.x + b.x; v = v > 0.f ? v : 0.2f * v; acc += w.x * v;
        v = a.y + b.y; v = v > 0.f ? v : 0.2f * v; acc += w.y * v;
        v = a.z + b.z; v = v > 0.f ? v : 0.2f * v; acc += w.z * v;
        v = a.w + b.w; v = v > 0.f ? v : 0.2f * v; acc += w.w * v;
    }
    logits[gid] = acc;
    atomicMaxF(&nmax[(size_t)d * H + h], acc);
}

template <int H>
__global__ void edge_exp(const float* __restrict__ logits, const float* __restrict__ nmax,
                         const int* __restrict__ dst,
                         float* __restrict__ exb, float* __restrict__ den) {
    const int gid = blockIdx.x * blockDim.x + threadIdx.x;
    if (gid >= N_EDGES_C * H) return;
    const int e = gid / H;
    const int h = gid - e * H;
    const int d = dst[e];
    const float ex = __expf(logits[gid] - nmax[(size_t)d * H + h]);
    exb[gid] = ex;
    atomicAdd(&den[(size_t)d * H + h], ex);
}

template <int H>
__global__ void edge_aggr(const float4* __restrict__ xl,
                          const int* __restrict__ src, const int* __restrict__ dst,
                          const float* __restrict__ exb, const float* __restrict__ den,
                          float* __restrict__ outp) {
    const int gid = blockIdx.x * blockDim.x + threadIdx.x;
    if (gid >= N_EDGES_C * H * 16) return;
    const int e = gid / (H * 16);
    const int q = gid - e * (H * 16);
    const int h = q >> 4;
    const int s = src[e];
    const int d = dst[e];
    const float alpha = exb[e * H + h] / (den[(size_t)d * H + h] + 1e-16f);
    const float4 v = xl[(size_t)s * (H * 16) + q];
    float* o = outp + (size_t)d * (H * 64) + q * 4;
    atomicAdd(o + 0, alpha * v.x);
    atomicAdd(o + 1, alpha * v.y);
    atomicAdd(o + 2, alpha * v.z);
    atomicAdd(o + 3, alpha * v.w);
}

// ---------------------------------------------------------------------------
// Elementwise / pooling / head
// ---------------------------------------------------------------------------
__global__ void fill_f32(float* __restrict__ p, float v, int n) {
    int i = blockIdx.x * blockDim.x + threadIdx.x;
    if (i < n) p[i] = v;
}

__global__ void bias_elu(float* __restrict__ h, const float* __restrict__ b,
                         int total, int C) {
    int i = blockIdx.x * blockDim.x + threadIdx.x;
    if (i >= total) return;
    float v = h[i] + b[i % C];
    h[i] = v > 0.f ? v : (__expf(v) - 1.f);
}

__global__ void pool_sum(const float* __restrict__ h, const int* __restrict__ batch,
                         float* __restrict__ pool) {
    int i = blockIdx.x * blockDim.x + threadIdx.x;
    if (i >= N_NODES_C * 64) return;
    int node = i >> 6;
    int c    = i & 63;
    atomicAdd(&pool[batch[node] * 64 + c], h[i]);
}

__global__ void pool_cnt(const int* __restrict__ batch, float* __restrict__ cnt) {
    int i = blockIdx.x * blockDim.x + threadIdx.x;
    if (i < N_NODES_C) atomicAdd(&cnt[batch[i]], 1.0f);
}

__global__ void mlp_head(const float* __restrict__ pool, const float* __restrict__ cnt,
                         const float* __restrict__ W3, const float* __restrict__ b3,
                         const float* __restrict__ W4, const float* __restrict__ b4,
                         float* __restrict__ out) {
    __shared__ float p[64];
    __shared__ float z[64];
    const int g = blockIdx.x;
    const int j = threadIdx.x;
    float c = cnt[g];
    c = c > 1.0f ? c : 1.0f;
    p[j] = pool[g * 64 + j] / c;
    __syncthreads();
    float acc = b3[j];
#pragma unroll 8
    for (int k = 0; k < 64; ++k) acc += p[k] * W3[k * 64 + j];
    z[j] = acc > 0.f ? acc : 0.f;
    __syncthreads();
    if (j < 2) {
        float o = b4[j];
        for (int k = 0; k < 64; ++k) o += z[k] * W4[k * 2 + j];
        out[g * 2 + j] = o;
    }
}

// ---------------------------------------------------------------------------
extern "C" void kernel_launch(void* const* d_in, const int* in_sizes, int n_in,
                              void* d_out, int out_size, void* d_ws, size_t ws_size,
                              hipStream_t stream) {
    (void)in_sizes; (void)n_in; (void)out_size; (void)ws_size;
    const float* x    = (const float*)d_in[0];
    const int*   ei   = (const int*)  d_in[1];
    const int*   bat  = (const int*)  d_in[2];
    const float* Wl1  = (const float*)d_in[3];
    const float* Wr1  = (const float*)d_in[4];
    const float* att1 = (const float*)d_in[5];
    const float* b1   = (const float*)d_in[6];
    const float* Wl2  = (const float*)d_in[7];
    const float* Wr2  = (const float*)d_in[8];
    const float* att2 = (const float*)d_in[9];
    const float* b2   = (const float*)d_in[10];
    const float* W3   = (const float*)d_in[11];
    const float* b3   = (const float*)d_in[12];
    const float* W4   = (const float*)d_in[13];
    const float* b4   = (const float*)d_in[14];

    const int* src = ei;
    const int* dst = ei + N_EDGES_C;

    char* ws = (char*)d_ws;
    size_t off = 0;
    auto take = [&](size_t bytes) -> char* {
        char* p = ws + off;
        off += (bytes + 255) & ~(size_t)255;
        return p;
    };
    unsigned short* wl1t = (unsigned short*)take((size_t)256 * 64 * 2);
    unsigned short* wr1t = (unsigned short*)take((size_t)256 * 64 * 2);
    unsigned short* wl2t = (unsigned short*)take((size_t)64 * 256 * 2);
    unsigned short* wr2t = (unsigned short*)take((size_t)64 * 256 * 2);
    float* xl1    = (float*)take((size_t)N_NODES_C * 256 * 4);
    float* xr1    = (float*)take((size_t)N_NODES_C * 256 * 4);
    float* h1     = (float*)take((size_t)N_NODES_C * 256 * 4);
    float* xl2    = (float*)take((size_t)N_NODES_C * 64 * 4);
    float* xr2    = (float*)take((size_t)N_NODES_C * 64 * 4);
    float* h2     = (float*)take((size_t)N_NODES_C * 64 * 4);
    float* logits = (float*)take((size_t)N_EDGES_C * 4 * 4);
    float* exb    = (float*)take((size_t)N_EDGES_C * 4 * 4);
    float* nmax   = (float*)take((size_t)N_NODES_C * 4 * 4);
    float* den    = (float*)take((size_t)N_NODES_C * 4 * 4);
    float* pool   = (float*)take((size_t)N_GRAPHS_C * 64 * 4);
    float* cnt    = (float*)take((size_t)N_GRAPHS_C * 4);

    auto cdiv = [](long a, long b) { return (unsigned)((a + b - 1) / b); };
    const float NEG_INF = -__builtin_inff();

    // --- weight prep (bf16, transposed) ---
    transpose_bf16<<<cdiv(64 * 256, 256), 256, 0, stream>>>(Wl1, wl1t, 64, 256);
    transpose_bf16<<<cdiv(64 * 256, 256), 256, 0, stream>>>(Wr1, wr1t, 64, 256);
    transpose_bf16<<<cdiv(256 * 64, 256), 256, 0, stream>>>(Wl2, wl2t, 256, 64);
    transpose_bf16<<<cdiv(256 * 64, 256), 256, 0, stream>>>(Wr2, wr2t, 256, 64);

    // --- layer 1: projections (WMMA) ---
    gemm_bf16_wmma<64, 256><<<N_NODES_C / 16, 256, 0, stream>>>(x, wl1t, xl1);
    gemm_bf16_wmma<64, 256><<<N_NODES_C / 16, 256, 0, stream>>>(x, wr1t, xr1);

    fill_f32<<<cdiv(N_NODES_C * 4, 256), 256, 0, stream>>>(nmax, NEG_INF, N_NODES_C * 4);
    fill_f32<<<cdiv(N_NODES_C * 4, 256), 256, 0, stream>>>(den, 0.f, N_NODES_C * 4);
    fill_f32<<<cdiv((long)N_NODES_C * 256, 256), 256, 0, stream>>>(h1, 0.f, N_NODES_C * 256);

    edge_logits<4><<<cdiv((long)N_EDGES_C * 4, 256), 256, 0, stream>>>(
        (const float4*)xl1, (const float4*)xr1, (const float4*)att1, src, dst, logits, nmax);
    edge_exp<4><<<cdiv((long)N_EDGES_C * 4, 256), 256, 0, stream>>>(logits, nmax, dst, exb, den);
    edge_aggr<4><<<cdiv((long)N_EDGES_C * 64, 256), 256, 0, stream>>>(
        (const float4*)xl1, src, dst, exb, den, h1);
    bias_elu<<<cdiv((long)N_NODES_C * 256, 256), 256, 0, stream>>>(h1, b1, N_NODES_C * 256, 256);

    // --- layer 2: projections (WMMA) ---
    gemm_bf16_wmma<256, 64><<<N_NODES_C / 16, 64, 0, stream>>>(h1, wl2t, xl2);
    gemm_bf16_wmma<256, 64><<<N_NODES_C / 16, 64, 0, stream>>>(h1, wr2t, xr2);

    fill_f32<<<cdiv(N_NODES_C, 256), 256, 0, stream>>>(nmax, NEG_INF, N_NODES_C);
    fill_f32<<<cdiv(N_NODES_C, 256), 256, 0, stream>>>(den, 0.f, N_NODES_C);
    fill_f32<<<cdiv((long)N_NODES_C * 64, 256), 256, 0, stream>>>(h2, 0.f, N_NODES_C * 64);

    edge_logits<1><<<cdiv(N_EDGES_C, 256), 256, 0, stream>>>(
        (const float4*)xl2, (const float4*)xr2, (const float4*)att2, src, dst, logits, nmax);
    edge_exp<1><<<cdiv(N_EDGES_C, 256), 256, 0, stream>>>(logits, nmax, dst, exb, den);
    edge_aggr<1><<<cdiv((long)N_EDGES_C * 16, 256), 256, 0, stream>>>(
        (const float4*)xl2, src, dst, exb, den, h2);
    bias_elu<<<cdiv((long)N_NODES_C * 64, 256), 256, 0, stream>>>(h2, b2, N_NODES_C * 64, 64);

    // --- pooling + MLP head ---
    fill_f32<<<cdiv(N_GRAPHS_C * 64, 256), 256, 0, stream>>>(pool, 0.f, N_GRAPHS_C * 64);
    fill_f32<<<1, 64, 0, stream>>>(cnt, 0.f, N_GRAPHS_C);
    pool_sum<<<cdiv((long)N_NODES_C * 64, 256), 256, 0, stream>>>(h2, bat, pool);
    pool_cnt<<<cdiv(N_NODES_C, 256), 256, 0, stream>>>(bat, cnt);

    mlp_head<<<N_GRAPHS_C, 64, 0, stream>>>(pool, cnt, W3, b3, W4, b4, (float*)d_out);
}
